// NearestUpsampling_26276609917176
// MI455X (gfx1250) — matrix operand order repaired and checked by reading, not compiled, not run on hardware
//
#include <hip/hip_runtime.h>

typedef __attribute__((ext_vector_type(16))) _Float16 v16h;
typedef __attribute__((ext_vector_type(8)))  _Float16 v8h;
typedef __attribute__((ext_vector_type(2)))  _Float16 v2h;
typedef __attribute__((ext_vector_type(8)))  float    v8f;

#define CIN    128
#define COUT   128
#define HDIM   64
#define WDIM   64
#define OHD    126
#define OWD    126
#define KDIM   512      // k = dx*256 + dy*128 + ci
#define NPITCH 264      // halves per n-row in LDS: 256 + 8 pad (528B, 16B-aligned)

// Folded 2x2 tap membership (nearest x2 upsample folds the 3x3):
// even parity: d=0 <- taps {0,1}, d=1 <- tap {2}
// odd  parity: d=0 <- tap {0},   d=1 <- taps {1,2}
__device__ __forceinline__ bool in_set(int par, int d, int t) {
  return par == 0 ? (d == 0 ? (t <= 1) : (t == 2))
                  : (d == 0 ? (t == 0) : (t >= 1));
}

// W2[p][q][cout][k], k = dx*256 + dy*128 + ci  (fp16, 512 KB)
__global__ __launch_bounds__(256) void fold_weights(const float* __restrict__ w,
                                                    _Float16* __restrict__ w2) {
  int idx  = blockIdx.x * 256 + threadIdx.x;    // 4*128*512 = 262144
  int kf   = idx & 511;
  int cout = (idx >> 9) & 127;
  int pq   = idx >> 16;
  int p = pq >> 1, q = pq & 1;
  int dx = kf >> 8, dy = (kf >> 7) & 1, ci = kf & 127;
  const float* wb = w + (size_t)((cout * CIN) + ci) * 9;
  float s = 0.f;
#pragma unroll
  for (int kh = 0; kh < 3; ++kh)
#pragma unroll
    for (int kw = 0; kw < 3; ++kw)
      if (in_set(p, dy, kh) && in_set(q, dx, kw))
        s += wb[kh * 3 + kw];
  w2[idx] = (_Float16)s;
}

// G[b][n][r][ci] = (fp16) x[b][ci][r][n]  -- 32x32 LDS-tiled transpose,
// coalesced reads (n-contiguous) and packed v2h coalesced writes (ci-contiguous).
__global__ __launch_bounds__(256) void transpose_x(const float* __restrict__ x,
                                                   _Float16* __restrict__ G) {
  __shared__ _Float16 tile[32][34];             // +4B row pad
  const int t  = threadIdx.x;
  const int tx = t & 31, ty = t >> 5;           // 32 x 8
  const int cb = blockIdx.x & 3;                // ci block (4 x 32)
  const int nb = (blockIdx.x >> 2) & 1;         // n  block (2 x 32)
  const int br = blockIdx.x >> 3;               // b*64 + r
  const int r = br & 63, b = br >> 6;
  const int ci0 = cb * 32, n0 = nb * 32;

  const float* xp = x + (((size_t)b * CIN + ci0) * HDIM + r) * WDIM + n0;
#pragma unroll
  for (int l = 0; l < 4; ++l)
    tile[ty + l * 8][tx] = (_Float16)xp[(size_t)(ty + l * 8) * HDIM * WDIM + tx];
  __syncthreads();

  const int ch = (t & 15) * 2;                  // even ci offset, v2h packed
  const int nl = t >> 4;                        // 0..15
  _Float16* Gp = G + (((size_t)b * 64 + n0) * 64 + r) * 128 + ci0;
#pragma unroll
  for (int l2 = 0; l2 < 2; ++l2) {
    int nn = nl + l2 * 16;
    v2h pk;
    pk[0] = tile[ch][nn];
    pk[1] = tile[ch + 1][nn];
    *(v2h*)(Gp + (size_t)nn * 64 * 128 + ch) = pk;
  }
}

// One block per (batch, source-row-pair r0): both output rows 2r0, 2r0+1, both
// column parities, 128 Couts x 63 cols from ONE staged tile.
// Staging: 65 contiguous 512B row copies G -> LDS (dy-outer k-order makes the
// (r0, r0+1) channel pair one contiguous run). 8 waves = 4 (p,q) GEMMs
// M=128 N=64 K=512; each wave 4 M-tiles x 4 N-tiles -> 1.0 loads/wmma.
__global__ __launch_bounds__(256) void upconv_wmma(const _Float16* __restrict__ G,
                                                   const _Float16* __restrict__ w2,
                                                   const float* __restrict__ bias,
                                                   float* __restrict__ out) {
  __shared__ _Float16 Bs[65 * NPITCH];          // ~33.5 KB

  const int t  = threadIdx.x;
  const int r0 = blockIdx.x;                    // 0..62 (rows r0, r0+1)
  const int b  = blockIdx.y;

  // ---- stage: Bs row n = G[b][n][r0][0..127] ++ G[b][n][r0+1][0..127] ----
  const _Float16* Gb = G + (size_t)b * 64 * 64 * 128;
  for (int i = t; i < 65 * 32; i += 256) {      // 65 rows x 32 pieces of 16B
    int n = i >> 5;
    int c = i & 31;
    int ncl = n > 63 ? 63 : n;                  // n==64 feeds only invalid cols
    *(v8h*)(&Bs[n * NPITCH + c * 8]) =
        *(const v8h*)(Gb + ((size_t)ncl * 64 + r0) * 128 + c * 8);
  }
  __syncthreads();

  // ---- WMMA: wave -> one (p,q); 64 Couts (4 M-tiles) x 64 cols (4 N-tiles) ----
  const int lane = t & 31;
  const int wave = t >> 5;
  const int n    = lane & 15;
  const int hi   = lane >> 4;
  const int pq   = wave >> 1;                   // (p<<1)|q
  const int p    = pq >> 1;
  const int q    = pq & 1;
  const int mb   = (wave & 1) * 64;

  // fp16 A 16x32 lane layout: lanes 0-15 K octets {0..7,16..23}, 16-31 {8..15,24..31}
  const _Float16* Abase = w2 + (size_t)(pq * COUT + mb + n) * KDIM + (hi << 3);

  v8f acc[16] = {{},{},{},{},{},{},{},{},{},{},{},{},{},{},{},{}};
#pragma unroll
  for (int kc = 0; kc < 16; ++kc) {
    const int sh = kc >> 3;                     // dx: +1 n-row shift in LDS
    const int kb = (kc & 7) * 32 + hi * 8;      // B: lanes 0-15 K {0..7,16..23}, 16-31 {8..15,24..31}
    union { v16h v; v8h h[2]; } a[4], bm[4];
#pragma unroll
    for (int mt = 0; mt < 4; ++mt) {
      const _Float16* ap = Abase + (size_t)(mt * 16) * KDIM + kc * 32;
      a[mt].h[0] = *(const v8h*)(ap);
      a[mt].h[1] = *(const v8h*)(ap + 16);
    }
#pragma unroll
    for (int nt = 0; nt < 4; ++nt) {
      const _Float16* bp = &Bs[(nt * 16 + n + sh) * NPITCH + kb];
      bm[nt].h[0] = *(const v8h*)(bp);
      bm[nt].h[1] = *(const v8h*)(bp + 16);
    }
#pragma unroll
    for (int mt = 0; mt < 4; ++mt)
#pragma unroll
      for (int nt = 0; nt < 4; ++nt)
        acc[mt * 4 + nt] = __builtin_amdgcn_wmma_f32_16x16x32_f16(
            false, a[mt].v, false, bm[nt].v, (short)0, acc[mt * 4 + nt],
            false, false);
  }

  // ---- store D (VGPR i: lanes0-15 M=i, lanes16-31 M=i+8) + bias ----
  const int oh = 2 * r0 + p;
#pragma unroll
  for (int mt = 0; mt < 4; ++mt) {
    float bv[8];
#pragma unroll
    for (int i = 0; i < 8; ++i) bv[i] = bias[mb + mt * 16 + hi * 8 + i];
#pragma unroll
    for (int nt = 0; nt < 4; ++nt) {
      int j  = nt * 16 + n;
      int ow = 2 * j + q;
      if (ow < OWD) {
        size_t obase =
            (((size_t)b * COUT + mb + mt * 16 + hi * 8) * OHD + oh) * OWD + ow;
#pragma unroll
        for (int i = 0; i < 8; ++i)
          out[obase + (size_t)i * OHD * OWD] = acc[mt * 4 + nt][i] + bv[i];
      }
    }
  }
}

extern "C" void kernel_launch(void* const* d_in, const int* in_sizes, int n_in,
                              void* d_out, int out_size, void* d_ws, size_t ws_size,
                              hipStream_t stream) {
  const float* x    = (const float*)d_in[0];   // [16,128,64,64]
  const float* w    = (const float*)d_in[1];   // [128,128,3,3]
  const float* bias = (const float*)d_in[2];   // [128]
  float* out = (float*)d_out;                  // [16,128,126,126]

  _Float16* w2 = (_Float16*)d_ws;                          // 512 KB
  _Float16* G  = (_Float16*)((char*)d_ws + (size_t)4 * COUT * KDIM * 2);  // 16.8 MB

  fold_weights<<<(4 * COUT * KDIM) / 256, 256, 0, stream>>>(w, w2);
  transpose_x<<<4 * 2 * 16 * 64, 256, 0, stream>>>(x, G);

  dim3 grid(63 /* source row pair */, 16 /* batch */);
  upconv_wmma<<<grid, 256, 0, stream>>>(G, w2, bias, out);
}
